// Autograd4bitQuantLinear_79293686219448
// MI455X (gfx1250) — compile-verified, matrix operand order checked by reading, not run on hardware
//
#include <hip/hip_runtime.h>

// GPTQ int4 dequant + GEMM for gfx1250 (MI455X), bf16 WMMA path.
//
// out[8192,4096] = x[8192,4096] @ W[4096,4096] + bias
//   W[k,n] = scales[k/128, n] * (nib4(qweight[k/8, n], k%8) - (nib4(qzeros[k/128, n/8], n%8) + 1))
//
// Block tile 256m x 128n x 32k, 8 waves (4m x 2n), each wave: 64x64 = 16 WMMAs/step.
// Tall tile amortizes int4-dequant VALU over 2x the matrix ops -> XDL-bound.

typedef __bf16 bf16;
typedef __attribute__((ext_vector_type(16))) __bf16 v16bf;
typedef __attribute__((ext_vector_type(8)))  __bf16 v8bf;
typedef __attribute__((ext_vector_type(4)))  __bf16 v4bf;
typedef __attribute__((ext_vector_type(8)))  float  v8f;
typedef __attribute__((ext_vector_type(4)))  float  v4f;

#define TOKENS 8192
#define NOUT   4096
#define KIN    4096
#define BM 256
#define BN 128
#define BK 32
#define STEPS (KIN / BK)       // 128 K-steps
#define ROWB   80              // LDS row: 32 bf16 (64B) + 16B pad -> conflict-free b128 frag loads
#define ATILEB (BM * ROWB)     // 20480 B
#define BTILEB (BN * ROWB)     // 10240 B
#define BUFB   (ATILEB + BTILEB)

union Frag { v16bf v; v8bf h[2]; };

__global__ __launch_bounds__(256)
void gptq_wmma_kernel(const float* __restrict__ x,
                      const int*   __restrict__ qweight,
                      const int*   __restrict__ qzeros,
                      const float* __restrict__ scales,
                      const float* __restrict__ bias,
                      float* __restrict__ out)
{
  __shared__ __align__(16) unsigned char smem[2 * BUFB];  // 60 KB, double buffered

  const int tid  = threadIdx.x;
  const int lane = tid & 31;
  const int wave = tid >> 5;
  const int m0 = blockIdx.y * BM;
  const int n0 = blockIdx.x * BN;

  // 8 waves -> 4 (m) x 2 (n); each wave owns 64m x 64n = 4x4 WMMA tiles.
  const int wm = (wave & 3) * 64;
  const int wn = (wave >> 2) * 64;

  v8f acc[4][4] = {};

  // B-dequant mapping: thread owns column bcol, two of four packed-k rows.
  const int bcol = tid & 127;
  const int k8a  = tid >> 7;           // 0 or 1 (also handles +2)
  const int ncol = n0 + bcol;
  float s = 0.0f, c = 0.0f;
  int cur_g = -1;

  auto stage = [&](int step) {
    const int k0 = step * BK;
    unsigned char* base = smem + (size_t)(step & 1) * BUFB;
    unsigned char* la = base;           // A tile: [m][32 bf16], 80B rows
    unsigned char* lb = base + ATILEB;  // B tile: [n][32 bf16], 80B rows (k-contiguous)

    // ---- A: 256x32 f32 -> bf16. 2048 float4s, 8 per thread, coalesced. ----
    #pragma unroll
    for (int i = 0; i < 8; ++i) {
      const int j   = i * 256 + tid;
      const int row = j >> 3;
      const int kq  = j & 7;
      const float* xp = x + (size_t)(m0 + row) * KIN + k0 + kq * 4;
      const v4f f = *(const v4f*)xp;
      __builtin_prefetch(xp + 2 * BK, 0, 1);   // global_prefetch_b8, 2 K-steps ahead
      v4bf h;
      h.x = (bf16)f.x; h.y = (bf16)f.y; h.z = (bf16)f.z; h.w = (bf16)f.w;
      *(v4bf*)(la + (size_t)row * ROWB + kq * 8) = h;
    }

    // ---- B: dequant 4x128 packed int32 -> 32x128 bf16 (stored n-major). ----
    const int g = step >> 2;            // group = k / 128 (uniform across wave)
    if (g != cur_g) {
      cur_g = g;
      s = scales[(size_t)g * NOUT + ncol];
      const int zw = qzeros[(size_t)g * (NOUT / 8) + (ncol >> 3)];
      const int z  = ((zw >> ((ncol & 7) * 4)) & 0xF) + 1;
      c = -s * (float)z;                // w = nib * s + c
    }
    const int k8base = k0 >> 3;
    #pragma unroll
    for (int r = 0; r < 2; ++r) {
      const int k8sub = k8a + r * 2;    // 0..3 within tile
      const int w = qweight[(size_t)(k8base + k8sub) * NOUT + ncol];
      v8bf h;
      #pragma unroll
      for (int jj = 0; jj < 8; ++jj) {
        const int nib = (w >> (jj * 4)) & 0xF;
        h[jj] = (bf16)fmaf((float)nib, s, c);
      }
      *(v8bf*)(lb + (size_t)bcol * ROWB + k8sub * 16) = h;   // ds_store_b128
    }
  };

  auto compute = [&](int step) {
    const unsigned char* base = smem + (size_t)(step & 1) * BUFB;
    const unsigned char* la = base;
    const unsigned char* lb = base + ATILEB;
    const int row   = lane & 15;   // M (for A) / N (for B)
    const int khalf = lane >> 4;   // K half select per ISA 16-bit fragment layout

    // Keep all 4 B fragments resident; stream A fragments one tile at a time.
    Frag b[4];
    #pragma unroll
    for (int tn = 0; tn < 4; ++tn) {
      const unsigned char* p = lb + (size_t)(wn + tn * 16 + row) * ROWB + khalf * 16;
      b[tn].h[0] = *(const v8bf*)p;         // elements 0..7  : K = khalf*8 .. +7
      b[tn].h[1] = *(const v8bf*)(p + 32);  // elements 8..15 : K = 16 + khalf*8 .. +7
    }
    #pragma unroll
    for (int tm = 0; tm < 4; ++tm) {
      Frag a;
      const unsigned char* p = la + (size_t)(wm + tm * 16 + row) * ROWB + khalf * 16;
      a.h[0] = *(const v8bf*)p;
      a.h[1] = *(const v8bf*)(p + 32);
      #pragma unroll
      for (int tn = 0; tn < 4; ++tn)
        acc[tm][tn] = __builtin_amdgcn_wmma_f32_16x16x32_bf16(
            false, a.v, false, b[tn].v, (short)0, acc[tm][tn], false, false);
    }
  };

  // Software pipeline: stage(k+1) overlaps compute(k); one barrier per step.
  stage(0);
  __syncthreads();
  for (int step = 0; step < STEPS; ++step) {
    if (step + 1 < STEPS) stage(step + 1);
    compute(step);
    __syncthreads();
  }

  // Epilogue: C/D layout -> lane holds (M = v + 8*(lane/16), N = lane%16) in VGPR v.
  const int nrow  = lane & 15;
  const int mhigh = (lane >> 4) * 8;
  #pragma unroll
  for (int tn = 0; tn < 4; ++tn) {
    const int col = n0 + wn + tn * 16 + nrow;
    const float bv = bias[col];
    #pragma unroll
    for (int tm = 0; tm < 4; ++tm) {
      const int rbase = m0 + wm + tm * 16 + mhigh;
      #pragma unroll
      for (int v = 0; v < 8; ++v)
        out[(size_t)(rbase + v) * NOUT + col] = acc[tm][tn][v] + bv;
    }
  }
}

extern "C" void kernel_launch(void* const* d_in, const int* in_sizes, int n_in,
                              void* d_out, int out_size, void* d_ws, size_t ws_size,
                              hipStream_t stream) {
  const float* x    = (const float*)d_in[0];  // [8192, 4096] f32
  const int*   qw   = (const int*)d_in[1];    // [512, 4096] int32 (8 nibbles along k)
  const int*   qz   = (const int*)d_in[2];    // [32, 512] int32 (8 nibbles along n)
  const float* sc   = (const float*)d_in[3];  // [32, 4096] f32
  // d_in[4] = g_idx, unused: g = k / 128 by construction
  const float* bias = (const float*)d_in[5];  // [4096] f32
  float* out = (float*)d_out;                 // [8192, 4096] f32

  dim3 grid(NOUT / BN, TOKENS / BM);          // (32, 32)
  gptq_wmma_kernel<<<grid, 256, 0, stream>>>(x, qw, qz, sc, bias, out);

  (void)in_sizes; (void)n_in; (void)out_size; (void)d_ws; (void)ws_size;
}
